// BAM_6253472383114
// MI455X (gfx1250) — matrix operand order repaired and checked by reading, not compiled, not run on hardware
//
#include <hip/hip_runtime.h>

#define DEVINL __device__ __forceinline__

typedef __bf16 bf16;
typedef __bf16 v16bf __attribute__((ext_vector_type(16)));
typedef float  v8f   __attribute__((ext_vector_type(8)));

DEVINL bf16 f2bf(float f) {
  unsigned int u = __builtin_bit_cast(unsigned int, f);
  u += 0x7fffu + ((u >> 16) & 1u);                  // round-to-nearest-even
  unsigned short h = (unsigned short)(u >> 16);
  return __builtin_bit_cast(bf16, h);
}

union FragBF {
  uint4 q[2];
  v16bf v;
};

// ---------------------------------------------------------------------------
// 1) 8x8 avg-pool: x[B,C,256,256] f32 -> xf[B,C,1024] bf16
// ---------------------------------------------------------------------------
__global__ __launch_bounds__(256) void bam_pool(const float* __restrict__ x,
                                                bf16* __restrict__ xf, int total) {
  int tid = blockIdx.x * 256 + threadIdx.x;
  if (tid >= total) return;
  int pw = tid & 31;
  int ph = (tid >> 5) & 31;
  long bc = (long)(tid >> 10);
  const float* base = x + bc * 65536 + (long)ph * 2048 + pw * 8;
  __builtin_prefetch(base + 4096, 0, 1);   // global_prefetch_b8: next row-pair
  float s = 0.f;
#pragma unroll
  for (int r = 0; r < 8; ++r) {
    float4 a = *(const float4*)(base + r * 256);
    float4 b = *(const float4*)(base + r * 256 + 4);
    s += (a.x + a.y) + (a.z + a.w) + (b.x + b.y) + (b.z + b.w);
  }
  xf[tid] = f2bf(s * (1.f / 64.f));
}

// ---------------------------------------------------------------------------
// 2) f32 -> bf16 convert (weights)
// ---------------------------------------------------------------------------
__global__ __launch_bounds__(256) void bam_cvt(const float* __restrict__ s,
                                               bf16* __restrict__ d, int n) {
  int i = blockIdx.x * 256 + threadIdx.x;
  if (i < n) d[i] = f2bf(s[i]);
}

// ---------------------------------------------------------------------------
// 3) Generic tiled WMMA bf16 GEMM: O = scale*(A x B) + bias
//    Block tile 64(M) x 256(N), 256 threads = 8 waves (2x4), wave = 32x64.
//    A tile staged with GLOBAL_LOAD_ASYNC_TO_LDS_B128 (ASYNCcnt),
//    B tile staged transposed so fragments load as ds_load_b128 pairs.
//    out_mode: 0 = f32 row-major, 1 = bf16 row-major, 2 = bf16 transposed
// ---------------------------------------------------------------------------
__global__ __launch_bounds__(256) void bam_gemm(
    const bf16* __restrict__ A,  long strideA, int lda,
    const bf16* __restrict__ Bm, long strideB, int ldb,
    float* __restrict__ outF, bf16* __restrict__ outB, long strideO, int ldo,
    const float* __restrict__ bias, int Kd, float scale, int out_mode) {
  __shared__ __align__(16) unsigned short lds_a[64][40];    // 64 x 32 bf16, pad to 40
  __shared__ __align__(16) unsigned short lds_b[256][40];   // B tile transposed: [n][k]

  const int b  = blockIdx.z;
  const int m0 = blockIdx.y * 64;
  const int n0 = blockIdx.x * 256;
  const int t    = threadIdx.x;
  const int lane = t & 31;
  const int w    = t >> 5;
  const int wm   = w & 1;    // 2 wave rows  (32 each)
  const int wn   = w >> 1;   // 4 wave cols  (64 each)
  const int half = lane >> 4;
  const int lm   = lane & 15;

  const bf16* Ab = A  + (long)b * strideA;
  const bf16* Bb = Bm + (long)b * strideB;

  v8f acc[2][4] = {};

  for (int k0 = 0; k0 < Kd; k0 += 32) {
    // async-stage A tile [64 x 32] row-major (one b128 per thread)
    {
      int r = t >> 2, c = (t & 3) * 8;
      unsigned lds_off = (unsigned)(size_t)(void*)&lds_a[r][c];
      unsigned long ga =
          (unsigned long)(const void*)(Ab + (long)(m0 + r) * lda + k0 + c);
      asm volatile("global_load_async_to_lds_b128 %0, %1, off"
                   :: "v"(lds_off), "v"(ga) : "memory");
    }
    // stage B tile [32 x 256] transposed into [256 x 32]
    {
      int kr = t >> 3, nc = (t & 7) * 8;
      const bf16* brow = Bb + (long)(k0 + kr) * ldb + n0;
#pragma unroll
      for (int rep = 0; rep < 4; ++rep) {
        int col = nc + rep * 64;
        uint4 d = *(const uint4*)(brow + col);
        const unsigned short* sp = (const unsigned short*)&d;
#pragma unroll
        for (int i = 0; i < 8; ++i) lds_b[col + i][kr] = sp[i];
      }
    }
    asm volatile("s_wait_asynccnt 0x0" ::: "memory");
    __syncthreads();

    FragBF af[2], bfr[4];
#pragma unroll
    for (int mt = 0; mt < 2; ++mt) {
      int row = wm * 32 + mt * 16 + lm;
      af[mt].q[0] = *(const uint4*)&lds_a[row][half * 8];       // K 0..7 / 8..15
      af[mt].q[1] = *(const uint4*)&lds_a[row][16 + half * 8];  // K 16..23 / 24..31
    }
#pragma unroll
    for (int nt = 0; nt < 4; ++nt) {
      int col = wn * 64 + nt * 16 + lm;
      bfr[nt].q[0] = *(const uint4*)&lds_b[col][half * 8];      // K 0..7 / 8..15
      bfr[nt].q[1] = *(const uint4*)&lds_b[col][16 + half * 8]; // K 16..23 / 24..31
    }
#pragma unroll
    for (int mt = 0; mt < 2; ++mt)
#pragma unroll
      for (int nt = 0; nt < 4; ++nt)
        acc[mt][nt] = __builtin_amdgcn_wmma_f32_16x16x32_bf16(
            false, af[mt].v, false, bfr[nt].v, (short)0, acc[mt][nt], false, false);
    __syncthreads();
  }

  // epilogue: C/D layout -> row = vgpr + 8*half, col = lane%16
#pragma unroll
  for (int mt = 0; mt < 2; ++mt) {
#pragma unroll
    for (int nt = 0; nt < 4; ++nt) {
      int mbase = m0 + wm * 32 + mt * 16;
      int nn = n0 + wn * 64 + nt * 16 + lm;
#pragma unroll
      for (int v = 0; v < 8; ++v) {
        int mm = mbase + v + half * 8;
        float val = acc[mt][nt][v] * scale + (bias ? bias[mm] : 0.f);
        if (out_mode == 0)
          outF[(long)b * strideO + (long)mm * ldo + nn] = val;
        else if (out_mode == 1)
          outB[(long)b * strideO + (long)mm * ldo + nn] = f2bf(val);
        else
          outB[(long)b * strideO + (long)nn * ldo + mm] = f2bf(val);
      }
    }
  }
}

// ---------------------------------------------------------------------------
// 4) Row softmax over energy[b,m,0..1023]; write attnT[b,n,m] in bf16
// ---------------------------------------------------------------------------
__global__ __launch_bounds__(256) void bam_softmax(const float* __restrict__ energy,
                                                   bf16* __restrict__ attnT) {
  const int N = 1024;
  int row = blockIdx.x;          // b*N + m
  int b = row >> 10;
  int m = row & 1023;
  const float* e = energy + (long)row * N;
  int t = threadIdx.x;

  float v0 = e[t], v1 = e[t + 256], v2 = e[t + 512], v3 = e[t + 768];
  float mx = fmaxf(fmaxf(v0, v1), fmaxf(v2, v3));

  __shared__ float red[256];
  red[t] = mx;
  __syncthreads();
  for (int s = 128; s > 0; s >>= 1) {
    if (t < s) red[t] = fmaxf(red[t], red[t + s]);
    __syncthreads();
  }
  float rmax = red[0];
  __syncthreads();

  v0 = __expf(v0 - rmax); v1 = __expf(v1 - rmax);
  v2 = __expf(v2 - rmax); v3 = __expf(v3 - rmax);
  red[t] = (v0 + v1) + (v2 + v3);
  __syncthreads();
  for (int s = 128; s > 0; s >>= 1) {
    if (t < s) red[t] += red[t + s];
    __syncthreads();
  }
  float inv = 1.f / red[0];

  bf16* o = attnT + (long)b * N * N + m;
  o[(long)t * N]         = f2bf(v0 * inv);
  o[(long)(t + 256) * N] = f2bf(v1 * inv);
  o[(long)(t + 512) * N] = f2bf(v2 * inv);
  o[(long)(t + 768) * N] = f2bf(v3 * inv);
}

// ---------------------------------------------------------------------------
// 5) nearest-upsample x8 + residual: out = repeat(out_small) + x
// ---------------------------------------------------------------------------
__global__ __launch_bounds__(256) void bam_upsample_add(const float* __restrict__ x,
                                                        const float* __restrict__ sm,
                                                        float* __restrict__ out) {
  long t = (long)blockIdx.x * 256 + threadIdx.x;  // one float4 per thread
  long idx = t * 4;
  int xw = (int)(idx & 255);
  int y  = (int)((idx >> 8) & 255);
  long bc = idx >> 16;
  float sv = sm[bc * 1024 + (y >> 3) * 32 + (xw >> 3)];
  float4 xi = *(const float4*)(x + idx);
  float4 o = make_float4(xi.x + sv, xi.y + sv, xi.z + sv, xi.w + sv);
  *(float4*)(out + idx) = o;
}

// ---------------------------------------------------------------------------
extern "C" void kernel_launch(void* const* d_in, const int* in_sizes, int n_in,
                              void* d_out, int out_size, void* d_ws, size_t ws_size,
                              hipStream_t stream) {
  const float* x  = (const float*)d_in[0];
  const float* Wq = (const float*)d_in[1];
  const float* bq = (const float*)d_in[2];
  const float* Wk = (const float*)d_in[3];
  const float* bk = (const float*)d_in[4];
  const float* Wv = (const float*)d_in[5];
  const float* bv = (const float*)d_in[6];
  float* out = (float*)d_out;

  const int B = 4, C = 512, N = 1024, K = 64;

  char* p = (char*)d_ws;
  auto carve = [&](size_t bytes) -> char* {
    char* r = p;
    p += (bytes + 255) & ~(size_t)255;
    return r;
  };
  bf16* xf     = (bf16*)carve((size_t)B * C * N * 2);   // pooled, bf16
  bf16* wqb    = (bf16*)carve((size_t)K * C * 2);
  bf16* wkb    = (bf16*)carve((size_t)K * C * 2);
  bf16* wvb    = (bf16*)carve((size_t)C * C * 2);
  bf16* qt     = (bf16*)carve((size_t)B * N * K * 2);   // q transposed [B,N,K]
  bf16* kmat   = (bf16*)carve((size_t)B * K * N * 2);   // [B,K,N]
  bf16* vmat   = (bf16*)carve((size_t)B * C * N * 2);   // [B,C,N]
  float* energy = (float*)carve((size_t)B * N * N * 4); // [B,N,N]
  bf16* attnT  = (bf16*)carve((size_t)B * N * N * 2);   // [B,N(n),N(m)]
  float* osm   = (float*)carve((size_t)B * C * N * 4);  // [B,C,N]

  // convert weights
  bam_cvt<<<(K * C + 255) / 256, 256, 0, stream>>>(Wq, wqb, K * C);
  bam_cvt<<<(K * C + 255) / 256, 256, 0, stream>>>(Wk, wkb, K * C);
  bam_cvt<<<(C * C + 255) / 256, 256, 0, stream>>>(Wv, wvb, C * C);

  // pool
  bam_pool<<<(B * C * N) / 256, 256, 0, stream>>>(x, xf, B * C * N);

  dim3 blk(256);
  // q = Wq x xf + bq  -> qt[b,n,k] (transposed bf16)
  bam_gemm<<<dim3(N / 256, 1, B), blk, 0, stream>>>(
      wqb, 0L, C, xf, (long)C * N, N,
      nullptr, qt, (long)N * K, K, bq, C, 1.f, 2);
  // k = Wk x xf + bk  -> kmat[b,k,n] (bf16)
  bam_gemm<<<dim3(N / 256, 1, B), blk, 0, stream>>>(
      wkb, 0L, C, xf, (long)C * N, N,
      nullptr, kmat, (long)K * N, N, bk, C, 1.f, 1);
  // v = Wv x xf + bv  -> vmat[b,c,n] (bf16)
  bam_gemm<<<dim3(N / 256, C / 64, B), blk, 0, stream>>>(
      wvb, 0L, C, xf, (long)C * N, N,
      nullptr, vmat, (long)C * N, N, bv, C, 1.f, 1);
  // energy = (qt x kmat) / sqrt(K) -> f32 [b,m,n]
  bam_gemm<<<dim3(N / 256, N / 64, B), blk, 0, stream>>>(
      qt, (long)N * K, K, kmat, (long)K * N, N,
      energy, nullptr, (long)N * N, N, nullptr, K, 0.125f, 0);
  // softmax rows -> attnT bf16
  bam_softmax<<<B * N, 256, 0, stream>>>(energy, attnT);
  // out_small = vmat x attnT -> f32 [b,c,m]
  bam_gemm<<<dim3(N / 256, C / 64, B), blk, 0, stream>>>(
      vmat, (long)C * N, N, attnT, (long)N * N, N,
      osm, nullptr, (long)C * N, N, nullptr, N, 1.f, 0);
  // upsample + residual
  bam_upsample_add<<<(B * C * 256 * 256 / 4) / 256, 256, 0, stream>>>(x, osm, out);
}
